// TokenSelect_smooth_1211180778201
// MI455X (gfx1250) — compile-verified, matrix operand order checked by reading, not run on hardware
//
#include <hip/hip_runtime.h>

typedef __attribute__((ext_vector_type(16))) _Float16 v16h;
typedef __attribute__((ext_vector_type(8)))  float    v8f;

#define BATCH 32
#define NTOK  4095      // tokens per batch (x[:,1:])
#define CH    768
#define KT    (CH / 32) // 24 K-tiles of 32
#define CHP   776       // padded LDS row stride (halves): 2-way-max ds conflicts
#define S0    614       // int(4095*0.15)
#define U0    3481      // 4095-614
#define M1    205       // 410//2
#define M2    205       // 410-205
#define S1    (S0 + M1) // 819
#define U1    (U0 - M1) // 3276
#define S2    (S1 + M2) // 1024
#define U2    (U1 - M2) // 3071
#define UPAD  3584
#define SPAD  1024
#define SORTN 4096
#define WPB   4              // waves per scores block
#define MROWS (WPB * 16)     // 64 unselect rows per block

union ABFrag { uint4 u[2]; v16h h; };

// ---------------------------------------------------------------- normalize
__global__ __launch_bounds__(256)
void norm_kernel(const float* __restrict__ x, _Float16* __restrict__ normTok) {
  const int row = blockIdx.x;             // b*NTOK + t
  const int b = row / NTOK, t = row % NTOK;
  const float* src = x + ((size_t)b * 4096 + t + 1) * CH;
  _Float16* dst = normTok + (size_t)row * CH;
  const int tid = threadIdx.x;
  float s = 0.f;
  for (int c = tid; c < CH; c += 256) { float v = src[c]; s += v * v; }
#pragma unroll
  for (int off = 16; off >= 1; off >>= 1) s += __shfl_xor(s, off, 32);
  __shared__ float red[8];
  __shared__ float tot;
  if ((tid & 31) == 0) red[tid >> 5] = s;
  __syncthreads();
  if (tid == 0) {
    float t2 = 0.f;
    for (int i = 0; i < 8; ++i) t2 += red[i];
    tot = t2;
  }
  __syncthreads();
  const float inv = rsqrtf(tot);
  for (int c = tid; c < CH; c += 256) dst[c] = (_Float16)(src[c] * inv);
}

// ---------------------------------------------------------------- init split
__global__ __launch_bounds__(256)
void init_idx_kernel(int* __restrict__ selIdx, int* __restrict__ unsel) {
  const int b = blockIdx.y;
  const int t = blockIdx.x * 256 + threadIdx.x;
  if (t >= NTOK) return;
  if ((t % 6 == 0) && (t / 6 < S0)) {
    selIdx[b * SPAD + t / 6] = t;
  } else {
    int f = (t + 5) / 6; if (f > S0) f = S0;   // #selected indices < t
    unsel[b * UPAD + (t - f)] = t;
  }
}

// -------- async staging: 12 x b128 per thread, LDS written by the async path.
// ISA: INST_OFFSET is added to BOTH the global and the LDS address, so one base
// VGPR pair + offset: immediates covers all chunks (no per-chunk address VALU).
#define ASYNC_LD(LDS, G, OFF)                                              \
  asm volatile("global_load_async_to_lds_b128 %0, %1, off offset:" OFF     \
               :: "v"(LDS), "v"(G) : "memory")

__device__ __forceinline__ void stage_async_tile(const char* gbase, unsigned ldsBase) {
  ASYNC_LD(ldsBase, gbase, "0");
  ASYNC_LD(ldsBase, gbase, "128");
  ASYNC_LD(ldsBase, gbase, "256");
  ASYNC_LD(ldsBase, gbase, "384");
  ASYNC_LD(ldsBase, gbase, "512");
  ASYNC_LD(ldsBase, gbase, "640");
  ASYNC_LD(ldsBase, gbase, "768");
  ASYNC_LD(ldsBase, gbase, "896");
  ASYNC_LD(ldsBase, gbase, "1024");
  ASYNC_LD(ldsBase, gbase, "1152");
  ASYNC_LD(ldsBase, gbase, "1280");
  ASYNC_LD(ldsBase, gbase, "1408");
}
__device__ __forceinline__ void wait_async_all() {
  asm volatile("s_wait_asynccnt 0x0" ::: "memory");
}

// ------------------------------------------- fused cosine GEMM + row max/argmax
// 4 waves/block, each wave owns a 16-row A strip (resident in VGPRs).
// B tiles double-buffered in LDS via GLOBAL_LOAD_ASYNC_TO_LDS_B128: tile nt+1
// streams into buf[1-cur] while the waves run WMMAs on buf[cur].
__global__ __launch_bounds__(128)
void scores_kernel(const _Float16* __restrict__ normTok,
                   const int* __restrict__ selIdx,
                   const int* __restrict__ unselIdx,
                   int S, int U,
                   float* __restrict__ nodeMax,
                   int* __restrict__ nodeArg,
                   int writeArg) {
  __shared__ _Float16 Bs[2][16][CHP];
  const int tid  = threadIdx.x;
  const int lane = tid & 31;
  const int wv   = tid >> 5;
  const int b = blockIdx.y;
  const int u0 = blockIdx.x * MROWS + wv * 16;
  const int l15  = lane & 15;
  const int koff = (lane < 16) ? 0 : 8;      // ISA 16-bit A/B fragment layout

  // gather this lane's A row; keep all 24 K-fragments in registers
  int urow = u0 + l15; if (urow > U - 1) urow = U - 1;
  const int garow = unselIdx[b * UPAD + urow];
  const _Float16* abase = normTok + ((size_t)b * NTOK + garow) * CH + koff;
  ABFrag a[KT];
#pragma unroll
  for (int kt = 0; kt < KT; ++kt) {
    a[kt].u[0] = *(const uint4*)(abase + kt * 32);
    a[kt].u[1] = *(const uint4*)(abase + kt * 32 + 16);
  }

  float best[8]; int bidx[8];
#pragma unroll
  for (int r = 0; r < 8; ++r) { best[r] = -__builtin_inff(); bidx[r] = 0; }

  // staging geometry: tile = 16 rows x 96 16B-chunks; 128 threads x 12 chunks.
  // thread covers row tid&15, chunks (tid>>4)+8j -> 128B-coalesced global reads.
  const int srow_local = tid & 15;
  const int cgrp16 = (tid >> 4) * 16;        // byte offset of chunk group
  // LDS base offsets (wave-relative LDS address = low 32 bits of generic ptr)
  const unsigned ldsRow0 = (unsigned)(size_t)&Bs[0][srow_local][0] + cgrp16;
  const unsigned ldsRow1 = (unsigned)(size_t)&Bs[1][srow_local][0] + cgrp16;

  const int ntiles = (S + 15) >> 4;

  // prologue: stage tile 0 into buffer 0
  {
    int scol = srow_local; if (scol > S - 1) scol = S - 1;
    const int brow = selIdx[b * SPAD + scol];
    stage_async_tile((const char*)(normTok + ((size_t)b * NTOK + brow) * CH) + cgrp16,
                     ldsRow0);
  }
  wait_async_all();
  __syncthreads();

  for (int nt = 0; nt < ntiles; ++nt) {
    const int cur = nt & 1;
    // stage tile nt+1 into the other buffer (no wait: overlaps with compute)
    if (nt + 1 < ntiles) {
      int scol = (nt + 1) * 16 + srow_local; if (scol > S - 1) scol = S - 1;
      const int brow = selIdx[b * SPAD + scol];
      stage_async_tile((const char*)(normTok + ((size_t)b * NTOK + brow) * CH) + cgrp16,
                       cur ? ldsRow0 : ldsRow1);
    }

    // compute: 24 WMMAs per wave from LDS B + register A
    v8f acc = {};
#pragma unroll
    for (int kt = 0; kt < KT; ++kt) {
      ABFrag bf;
      bf.u[0] = *(const uint4*)(&Bs[cur][l15][kt * 32 + koff]);
      bf.u[1] = *(const uint4*)(&Bs[cur][l15][kt * 32 + koff + 16]);
      acc = __builtin_amdgcn_wmma_f32_16x16x32_f16(
          false, a[kt].h, false, bf.h, (short)0, acc, false, false);
    }

    const int col = nt * 16 + l15;
    if (col < S) {                           // mask padded columns out of max
#pragma unroll
      for (int r = 0; r < 8; ++r)
        if (acc[r] > best[r]) { best[r] = acc[r]; bidx[r] = col; }
    }

    wait_async_all();                        // our nt+1 loads landed in LDS
    __syncthreads();                         // everyone done reading buf[cur]
  }

  // reduce across the 16 lanes holding one row's columns (rows 0-7: lanes 0-15,
  // rows 8-15: lanes 16-31); ties -> smaller index (argmax first-occurrence)
#pragma unroll
  for (int off = 8; off >= 1; off >>= 1) {
#pragma unroll
    for (int r = 0; r < 8; ++r) {
      float ov = __shfl_xor(best[r], off, 32);
      int   oi = __shfl_xor(bidx[r], off, 32);
      if (ov > best[r] || (ov == best[r] && oi < bidx[r])) { best[r] = ov; bidx[r] = oi; }
    }
  }

  if (lane == 0 || lane == 16) {
    const int rbase = u0 + ((lane == 16) ? 8 : 0);
#pragma unroll
    for (int r = 0; r < 8; ++r) {
      const int u = rbase + r;
      if (u < U) {
        nodeMax[b * UPAD + u] = best[r];
        if (writeArg) nodeArg[b * UPAD + u] = bidx[r];
      }
    }
  }
}

// ------------------------- stable argsort (bitonic on (value,index) keys) + update
__global__ __launch_bounds__(1024)
void sort_update_kernel(const float* __restrict__ nodeMax,
                        const int* __restrict__ unselSrc,
                        int* __restrict__ unselDst,
                        int* __restrict__ selIdx,
                        int U, int m, int S) {
  __shared__ unsigned long long keys[SORTN];
  const int tid = threadIdx.x;
  const int b = blockIdx.x;
  for (int i = tid; i < SORTN; i += 1024) {
    unsigned long long key;
    if (i < U) {
      unsigned uu = __float_as_uint(nodeMax[b * UPAD + i]);
      unsigned mask = (uu & 0x80000000u) ? 0xffffffffu : 0x80000000u;
      key = ((unsigned long long)(uu ^ mask) << 32) | (unsigned)i;
    } else {
      key = ~0ull;                       // padding sorts to the end
    }
    keys[i] = key;
  }
  for (int k = 2; k <= SORTN; k <<= 1)
    for (int j = k >> 1; j > 0; j >>= 1) {
      __syncthreads();
      for (int i = tid; i < SORTN; i += 1024) {
        int l = i ^ j;
        if (l > i) {
          bool asc = (i & k) == 0;
          unsigned long long a = keys[i], c = keys[l];
          if ((a > c) == asc) { keys[i] = c; keys[l] = a; }
        }
      }
    }
  __syncthreads();
  const int newU = U - m;
  for (int i = tid; i < newU; i += 1024) {   // unadd part, in sorted order
    int pos = (int)(unsigned)(keys[i] & 0xffffffffu);
    unselDst[b * UPAD + i] = unselSrc[b * UPAD + pos];
  }
  for (int i = tid; i < m; i += 1024) {      // top-m appended to select set
    int pos = (int)(unsigned)(keys[newU + i] & 0xffffffffu);
    selIdx[b * SPAD + S + i] = unselSrc[b * UPAD + pos];
  }
}

// ------------------------------- merge: atomic-free, deterministic (ascending-u)
#define CHUNK ((U2 + 255) / 256)   // 12 u's per thread
__global__ __launch_bounds__(256)
void merge_kernel(const float* __restrict__ x,
                  const int* __restrict__ selIdx,
                  const int* __restrict__ unselIdx,
                  const int* __restrict__ nodeArg,
                  float* __restrict__ out, float* __restrict__ idxOut) {
  const int s = blockIdx.x, b = blockIdx.y, tid = threadIdx.x;
  __shared__ int list[CHUNK * 256];      // token rows merging into s (u-ascending)
  __shared__ int csum[256];

  // per-thread contiguous chunk -> matches stay in ascending-u order
  int myu[CHUNK];
  int nloc = 0;
#pragma unroll
  for (int k = 0; k < CHUNK; ++k) {
    const int u = tid * CHUNK + k;
    if (u < U2 && nodeArg[b * UPAD + u] == s) myu[nloc++] = u;
  }
  csum[tid] = nloc;
  __syncthreads();
  // Hillis-Steele inclusive scan over per-thread counts
  for (int off = 1; off < 256; off <<= 1) {
    int v = (tid >= off) ? csum[tid - off] : 0;
    __syncthreads();
    csum[tid] += v;
    __syncthreads();
  }
  const int base = csum[tid] - nloc;
  const int total = csum[255];
  for (int k = 0; k < nloc; ++k)
    list[base + k] = unselIdx[b * UPAD + myu[k]];
  __syncthreads();

  const int grow = selIdx[b * SPAD + s] + 1;
  const float inv = 1.0f / (float)(1 + total);
  const float* srcSel = x + ((size_t)b * 4096 + grow) * CH;
  float* dst = out + ((size_t)b * S2 + s) * CH;
  for (int c = tid; c < CH; c += 256) {
    float v = srcSel[c];
    for (int i = 0; i < total; ++i)
      v += x[((size_t)b * 4096 + list[i] + 1) * CH + c];
    dst[c] = v * inv;
  }
  if (tid == 0) idxOut[b * SPAD + s] = (float)selIdx[b * SPAD + s];
}

// ---------------------------------------------------------------- launcher
extern "C" void kernel_launch(void* const* d_in, const int* in_sizes, int n_in,
                              void* d_out, int out_size, void* d_ws, size_t ws_size,
                              hipStream_t stream) {
  (void)in_sizes; (void)n_in; (void)out_size; (void)ws_size;
  const float* x = (const float*)d_in[0];
  float* out = (float*)d_out;

  char* ws = (char*)d_ws;
  size_t off = 0;
  _Float16* normTok = (_Float16*)(ws + off); off += (size_t)BATCH * NTOK * CH * sizeof(_Float16);
  float* nodeMax = (float*)(ws + off);       off += (size_t)BATCH * UPAD * sizeof(float);
  int*   nodeArg = (int*)(ws + off);         off += (size_t)BATCH * UPAD * sizeof(int);
  int*   selIdx  = (int*)(ws + off);         off += (size_t)BATCH * SPAD * sizeof(int);
  int*   unselA  = (int*)(ws + off);         off += (size_t)BATCH * UPAD * sizeof(int);
  int*   unselB  = (int*)(ws + off);         off += (size_t)BATCH * UPAD * sizeof(int);

  float* acc    = out;                           // (B, 1024, 768) select_token
  float* idxOut = out + (size_t)BATCH * S2 * CH; // (B, 1024) select_index (as float)

  norm_kernel<<<BATCH * NTOK, 256, 0, stream>>>(x, normTok);
  init_idx_kernel<<<dim3((NTOK + 255) / 256, BATCH), 256, 0, stream>>>(selIdx, unselA);

  // round 1
  scores_kernel<<<dim3((U0 + MROWS - 1) / MROWS, BATCH), 128, 0, stream>>>(
      normTok, selIdx, unselA, S0, U0, nodeMax, nodeArg, 0);
  sort_update_kernel<<<BATCH, 1024, 0, stream>>>(nodeMax, unselA, unselB, selIdx, U0, M1, S0);

  // round 2
  scores_kernel<<<dim3((U1 + MROWS - 1) / MROWS, BATCH), 128, 0, stream>>>(
      normTok, selIdx, unselB, S1, U1, nodeMax, nodeArg, 0);
  sort_update_kernel<<<BATCH, 1024, 0, stream>>>(nodeMax, unselB, unselA, selIdx, U1, M2, S1);

  // final argmax round
  scores_kernel<<<dim3((U2 + MROWS - 1) / MROWS, BATCH), 128, 0, stream>>>(
      normTok, selIdx, unselA, S2, U2, nodeMax, nodeArg, 1);

  // merge (single kernel, no atomics, deterministic accumulation order)
  merge_kernel<<<dim3(S2, BATCH), 256, 0, stream>>>(x, selIdx, unselA, nodeArg, acc, idxOut);
}
